// IntentionAwareAttention_32959579029608
// MI455X (gfx1250) — matrix operand order repaired and checked by reading, not compiled, not run on hardware
//
#include <hip/hip_runtime.h>

// MI455X / gfx1250: all matmuls via v_wmma_f32_16x16x32_bf16, flash-style
// attention, A/K fragment tiles staged with global_load_async_to_lds_b128
// (ASYNCcnt path), softmax reductions on VALU DPP (co-executes with WMMA).

typedef __bf16 bf16;
typedef __attribute__((ext_vector_type(16))) __bf16 v16bf;
typedef __attribute__((ext_vector_type(8)))  __bf16 v8bf;
typedef __attribute__((ext_vector_type(8)))  float  v8f;

__device__ __forceinline__ bf16 f2bf(float f) {
  unsigned int u = __builtin_bit_cast(unsigned int, f);
  u += 0x7FFFu + ((u >> 16) & 1u);          // round-to-nearest-even
  unsigned short h = (unsigned short)(u >> 16);
  return __builtin_bit_cast(bf16, h);
}

// ISA 7.12.2: 16-bit A 16x32 fragment. lane = h*16 + m, element e:
//   e<8 -> K = e + 8h ; e>=8 -> K = 16 + (e-8) + 8h
__device__ __forceinline__ int afrag_idx(int m, int k) {
  int h = (k >> 3) & 1;
  int e = (k & 7) + ((k >> 4) & 1) * 8;
  return (h * 16 + m) * 16 + e;
}
// 16-bit B 32x16 fragment: lane = ((k>>3)&1)*16 + n, element e = (k&7) + ((k>>4)&1)*8
__device__ __forceinline__ int bfrag_idx(int k, int n) {
  return (((k >> 3) & 1) * 16 + n) * 16 + (k & 7) + ((k >> 4) & 1) * 8;
}

__device__ __forceinline__ v8f wmma_bf16(v16bf a, v16bf b, v8f c) {
  return __builtin_amdgcn_wmma_f32_16x16x32_bf16(false, a, false, b, (short)0, c,
                                                 false, false);
}

// ---- CDNA5 async global->LDS (tracked by ASYNCcnt) -------------------------
// LDS aperture keeps the wave-relative byte offset in addr[31:0] (ISA 10.2),
// so truncating a generic __shared__ pointer yields the DS address operand.
__device__ __forceinline__ unsigned lds_off_u32(const void* p) {
  return (unsigned)(unsigned long long)p;
}
__device__ __forceinline__ void async_ld_b128(unsigned lds_off, const void* g) {
  asm volatile("global_load_async_to_lds_b128 %0, %1, off"
               :: "v"(lds_off), "v"((unsigned long long)g) : "memory");
}
__device__ __forceinline__ void wait_async0() {
  asm volatile("s_wait_asynccnt 0x0" ::: "memory");
}

// ---- VALU DPP row reductions (stay within 16-lane half-row on wave32) ------
template <int CTRL>
__device__ __forceinline__ float dppf(float x) {
  int i = __builtin_bit_cast(int, x);
  i = __builtin_amdgcn_update_dpp(0, i, CTRL, 0xF, 0xF, true);
  return __builtin_bit_cast(float, i);
}
__device__ __forceinline__ float row_max16(float t) {
  t = fmaxf(t, dppf<0xB1>(t));    // quad_perm xor1
  t = fmaxf(t, dppf<0x4E>(t));    // quad_perm xor2
  t = fmaxf(t, dppf<0x124>(t));   // row_ror:4
  t = fmaxf(t, dppf<0x128>(t));   // row_ror:8
  return t;
}
__device__ __forceinline__ float row_sum16(float t) {
  t += dppf<0xB1>(t);
  t += dppf<0x4E>(t);
  t += dppf<0x124>(t);
  t += dppf<0x128>(t);
  return t;
}

// ---------------------------------------------------------------- converts
__global__ __launch_bounds__(256) void cvt_bf16_k(const float* __restrict__ src,
                                                  bf16* __restrict__ dst,
                                                  int n, float scale) {
  int i = blockIdx.x * 256 + threadIdx.x;
  if (i < n) dst[i] = f2bf(src[i] * scale);
}

__global__ __launch_bounds__(256) void scale_f32_k(const float* __restrict__ src,
                                                   float* __restrict__ dst,
                                                   int n, float scale) {
  int i = blockIdx.x * 256 + threadIdx.x;
  if (i < n) dst[i] = src[i] * scale;
}

// ---------------------------------------------------------------- GEMM
// C(MxN) = A(MxK,bf16) @ B(KxN,bf16) + bias ; 128x128 block, 8 waves (2x4).
// A tile: per-lane fragment slots are contiguous 16B global runs -> staged
// straight into fragment order with global_load_async_to_lds_b128.
// B tile: needs transpose -> VALU scatter into fragment order.
template <bool OUT_BF16>
__global__ __launch_bounds__(256) void gemm_bf16_k(
    const bf16* __restrict__ A, const bf16* __restrict__ B,
    const float* __restrict__ bias, void* __restrict__ Cvoid,
    int M, int N, int K) {
  __shared__ bf16 sA[4096];   // 128x32 A tile, fragment order (8 KB)
  __shared__ bf16 sB[4096];   // 32x128 B tile, fragment order (8 KB)
  const int tid  = threadIdx.x;
  const int lane = tid & 31, wid = tid >> 5;
  const int wr = wid & 1, wc = wid >> 1;       // wave grid 2 (rows) x 4 (cols)
  const int hh = lane >> 4, nn = lane & 15;
  const int m0 = blockIdx.y * 128, n0 = blockIdx.x * 128;

  v8f zero = {};
  v8f acc[4][2];
#pragma unroll
  for (int i = 0; i < 4; ++i)
#pragma unroll
    for (int j = 0; j < 2; ++j) acc[i][j] = zero;

  const unsigned sA_lane = lds_off_u32(&sA[wid * 512 + lane * 16]);
  const bf16* Arow = A + (size_t)(m0 + wid * 16 + nn) * K + 8 * hh;

  for (int k0 = 0; k0 < K; k0 += 32) {
    // A: wave `wid` owns row-block `wid`; two 16B runs per lane, async to LDS
    async_ld_b128(sA_lane,      Arow + k0);        // K = 8h .. 8h+7
    async_ld_b128(sA_lane + 16, Arow + k0 + 16);   // K = 16+8h .. 16+8h+7
    // B: 32 k x 128 cols, transpose-scatter into fragment order
#pragma unroll
    for (int ch = tid; ch < 512; ch += 256) {
      int kk = ch >> 4, nc = (ch & 15) * 8;
      v8bf db = *(const v8bf*)(B + (size_t)(k0 + kk) * N + n0 + nc);
#pragma unroll
      for (int e = 0; e < 8; ++e)
        sB[((nc + e) >> 4) * 512 + bfrag_idx(kk, (nc + e) & 15)] = db[e];
    }
    if (k0 + 32 < K) {  // global_prefetch_b8 next tiles
      __builtin_prefetch(Arow + k0 + 32, 0, 1);
      __builtin_prefetch(B + (size_t)(k0 + 32 + (tid >> 4)) * N + n0 + (tid & 15) * 8, 0, 1);
    }
    wait_async0();
    __syncthreads();

    v16bf a[4], b2[2];
#pragma unroll
    for (int i = 0; i < 4; ++i)
      a[i] = *(const v16bf*)&sA[(wr * 4 + i) * 512 + lane * 16];
#pragma unroll
    for (int j = 0; j < 2; ++j)
      b2[j] = *(const v16bf*)&sB[(wc * 2 + j) * 512 + lane * 16];
#pragma unroll
    for (int i = 0; i < 4; ++i)
#pragma unroll
      for (int j = 0; j < 2; ++j)
        acc[i][j] = wmma_bf16(a[i], b2[j], acc[i][j]);
    __syncthreads();
  }

  bf16*  Cb = (bf16*)Cvoid;
  float* Cf = (float*)Cvoid;
#pragma unroll
  for (int i = 0; i < 4; ++i) {
#pragma unroll
    for (int j = 0; j < 2; ++j) {
      int col = n0 + wc * 32 + j * 16 + nn;
      float bv = bias[col];
#pragma unroll
      for (int r = 0; r < 8; ++r) {
        int row = m0 + wr * 64 + i * 16 + r + 8 * hh;  // C/D layout
        float v = acc[i][j][r] + bv;
        if (OUT_BF16) Cb[(size_t)row * N + col] = f2bf(v);
        else          Cf[(size_t)row * N + col] = v;
      }
    }
  }
}

// ---------------------------------------------------------------- flash attention
// Per (b,h): Qeff = contiguous Q[b, h*128:(h+1)*128, :] viewed as (1024 x 128)
// (reproduces the reference's no-transpose q.reshape(bs,H,L,Dh) exactly);
// K/V = head-h column slice of KV. Block = 4 waves x 16 eff-queries = 64 rows.
__global__ __launch_bounds__(128) void flash_attn_k(
    const bf16* __restrict__ Q, const bf16* __restrict__ KV,
    bf16* __restrict__ O) {
  __shared__ bf16 sK[4096];   // 32 keys x 128 dh, B-fragment order for S
  __shared__ bf16 sV[4096];   // 32 keys x 128 dh, B-fragment order for P@V
  __shared__ bf16 sP[2048];   // per-wave 16x32 P tile in A-fragment order
  const int tid  = threadIdx.x;
  const int lane = tid & 31, wid = tid >> 5;
  const int hh = lane >> 4, nn = lane & 15;
  const int b = blockIdx.y >> 3, h = blockIdx.y & 7;
  const int q0 = blockIdx.x * 64;
  const size_t qbase = (size_t)(b * 1024 + h * 128) * 1024;  // (1024x128) view

  // A-fragments of the wave's 16 queries, 4 chunks of K=32 over Dh=128.
  v16bf qf[4];
  {
    const bf16* qr = Q + qbase + (size_t)(q0 + wid * 16 + nn) * 128;
#pragma unroll
    for (int c = 0; c < 4; ++c) {
      union { v16bf v; v8bf p[2]; } u;
      u.p[0] = *(const v8bf*)(qr + c * 32 + 8 * hh);        // K = 0..7  (+8h)
      u.p[1] = *(const v8bf*)(qr + c * 32 + 16 + 8 * hh);   // K = 16..23(+8h)
      qf[c] = u.v;
    }
  }

  v8f zero = {};
  v8f acc[8];              // 16 rows x 128 dh output accumulator
  float mrow[8], lrow[8];  // per-lane row stats (rows r + 8*hh)
#pragma unroll
  for (int r = 0; r < 8; ++r) { mrow[r] = -3.0e38f; lrow[r] = 0.0f; }
#pragma unroll
  for (int d = 0; d < 8; ++d) acc[d] = zero;

  const size_t kvrow0 = (size_t)b * 2048 * 2048;
  const int kcol = h * 128;

  for (int n0 = 0; n0 < 2048; n0 += 32) {
    // K tile: fragment lane slots are two 16B global runs per lane
    // -> async copy straight into B-fragment order. Wave owns 2 blocks.
#pragma unroll
    for (int t = 0; t < 2; ++t) {
      int blk = wid * 2 + t;                 // (kb,c) = (blk>>2, blk&3)
      int kb = blk >> 2, c = blk & 3;
      const bf16* ga = KV + kvrow0 + (size_t)(n0 + kb * 16 + nn) * 2048 +
                       kcol + c * 32 + 8 * hh;
      unsigned lo = lds_off_u32(&sK[blk * 512 + lane * 16]);
      async_ld_b128(lo,      ga);        // dh = 8h .. 8h+7 of this chunk
      async_ld_b128(lo + 16, ga + 16);   // dh = 16+8h .. 16+8h+7
    }
    // V tile: needs key<->dh transpose -> VALU scatter into fragment order
#pragma unroll
    for (int ch = tid; ch < 512; ch += 128) {
      int key = ch >> 4, dhc = (ch & 15) * 8;
      v8bf vd = *(const v8bf*)(KV + kvrow0 + (size_t)(n0 + key) * 2048 +
                               kcol + 1024 + dhc);
#pragma unroll
      for (int e = 0; e < 8; ++e) {
        int dh = dhc + e;
        sV[(dh >> 4) * 512 + bfrag_idx(key, dh & 15)] = vd[e];
      }
    }
    wait_async0();
    __syncthreads();

    // S = Qeff @ K^T : two 16x16 score tiles (keys 0-15, 16-31)
    v8f s0 = zero, s1 = zero;
#pragma unroll
    for (int c = 0; c < 4; ++c) {
      v16bf kb0 = *(const v16bf*)&sK[(0 * 4 + c) * 512 + lane * 16];
      v16bf kb1 = *(const v16bf*)&sK[(1 * 4 + c) * 512 + lane * 16];
      s0 = wmma_bf16(qf[c], kb0, s0);
      s1 = wmma_bf16(qf[c], kb1, s1);
    }

    // online softmax over this 32-key slab; row reductions on VALU DPP
#pragma unroll
    for (int r = 0; r < 8; ++r) {
      float t = row_max16(fmaxf(s0[r], s1[r]));
      float nm = fmaxf(mrow[r], t);
      float sc = __expf(mrow[r] - nm);
      mrow[r] = nm;
      float p0 = __expf(s0[r] - nm);
      float p1 = __expf(s1[r] - nm);
      lrow[r] = lrow[r] * sc + row_sum16(p0 + p1);
#pragma unroll
      for (int d = 0; d < 8; ++d) acc[d][r] = acc[d][r] * sc;
      // scatter P into A-fragment layout (C/D -> A transpose via LDS)
      int m = r + 8 * hh;
      sP[wid * 512 + afrag_idx(m, nn)]      = f2bf(p0);
      sP[wid * 512 + afrag_idx(m, nn + 16)] = f2bf(p1);
    }

    v16bf pa = *(const v16bf*)&sP[wid * 512 + lane * 16];
#pragma unroll
    for (int d = 0; d < 8; ++d) {
      v16bf vf = *(const v16bf*)&sV[d * 512 + lane * 16];
      acc[d] = wmma_bf16(pa, vf, acc[d]);
    }
    __syncthreads();
  }

  // O (same (1024x128) contiguous view as Qeff)
#pragma unroll
  for (int d = 0; d < 8; ++d) {
#pragma unroll
    for (int r = 0; r < 8; ++r) {
      int qrow = q0 + wid * 16 + r + 8 * hh;
      O[qbase + (size_t)qrow * 128 + d * 16 + nn] = f2bf(acc[d][r] / lrow[r]);
    }
  }
}

// ---------------------------------------------------------------- launch
extern "C" void kernel_launch(void* const* d_in, const int* in_sizes, int n_in,
                              void* d_out, int out_size, void* d_ws, size_t ws_size,
                              hipStream_t stream) {
  (void)in_sizes; (void)n_in; (void)out_size; (void)ws_size;
  const float* motion = (const float*)d_in[0];
  const float* scene  = (const float*)d_in[1];
  const float* Wq     = (const float*)d_in[2];
  const float* bq     = (const float*)d_in[3];
  const float* Wkv    = (const float*)d_in[4];
  const float* bkv    = (const float*)d_in[5];
  const float* Wout   = (const float*)d_in[6];
  const float* bout   = (const float*)d_in[7];
  float* out = (float*)d_out;

  const size_t nMotion = (size_t)8 * 1024 * 1024;
  const size_t nScene  = (size_t)8 * 2048 * 1024;
  const size_t nWq = (size_t)1024 * 1024;
  const size_t nWkv = (size_t)1024 * 2048;
  const size_t nWout = (size_t)1024 * 1024;
  const size_t nQ  = (size_t)8192 * 1024;
  const size_t nKV = (size_t)16384 * 2048;
  const size_t nO  = (size_t)8192 * 1024;

  bf16* motion_bf = (bf16*)d_ws;
  bf16* scene_bf  = motion_bf + nMotion;
  bf16* Wq_bf     = scene_bf + nScene;     // scaled by D^-0.5
  bf16* Wkv_bf    = Wq_bf + nWq;
  bf16* Wout_bf   = Wkv_bf + nWkv;
  bf16* Qbuf      = Wout_bf + nWout;
  bf16* KVbuf     = Qbuf + nQ;
  bf16* Obuf      = KVbuf + nKV;
  float* bq_s     = (float*)(Obuf + nO);   // total d_ws use ~159.4 MB

  const float s = 0.03125f;  // 1024^-0.5

  cvt_bf16_k<<<(unsigned)((nMotion + 255) / 256), 256, 0, stream>>>(motion, motion_bf, (int)nMotion, 1.0f);
  cvt_bf16_k<<<(unsigned)((nScene + 255) / 256), 256, 0, stream>>>(scene, scene_bf, (int)nScene, 1.0f);
  cvt_bf16_k<<<(unsigned)((nWq + 255) / 256), 256, 0, stream>>>(Wq, Wq_bf, (int)nWq, s);
  cvt_bf16_k<<<(unsigned)((nWkv + 255) / 256), 256, 0, stream>>>(Wkv, Wkv_bf, (int)nWkv, 1.0f);
  cvt_bf16_k<<<(unsigned)((nWout + 255) / 256), 256, 0, stream>>>(Wout, Wout_bf, (int)nWout, 1.0f);
  scale_f32_k<<<4, 256, 0, stream>>>(bq, bq_s, 1024, s);

  // Q = motion @ (Wq*s) + bq*s   -> bf16 (8192 x 1024)
  gemm_bf16_k<true><<<dim3(1024 / 128, 8192 / 128), 256, 0, stream>>>(
      motion_bf, Wq_bf, bq_s, Qbuf, 8192, 1024, 1024);
  // KV = scene @ Wkv + bkv       -> bf16 (16384 x 2048), [:,0:1024]=K, [:,1024:]=V
  gemm_bf16_k<true><<<dim3(2048 / 128, 16384 / 128), 256, 0, stream>>>(
      scene_bf, Wkv_bf, bkv, KVbuf, 16384, 2048, 1024);
  // attention (flash, online softmax over N=2048)
  flash_attn_k<<<dim3(16, 64), 128, 0, stream>>>(Qbuf, KVbuf, Obuf);
  // final = O @ Wout + bout      -> f32 (8192 x 1024)
  gemm_bf16_k<false><<<dim3(1024 / 128, 8192 / 128), 256, 0, stream>>>(
      Obuf, Wout_bf, bout, out, 8192, 1024, 1024);
}
